// FeedbackModel_74311524155787
// MI455X (gfx1250) — compile-verified, matrix operand order checked by reading, not run on hardware
//
#include <hip/hip_runtime.h>
#include <hip/hip_bf16.h>

typedef __attribute__((ext_vector_type(16))) _Float16 v16h;
typedef __attribute__((ext_vector_type(8)))  _Float16 v8h;
typedef __attribute__((ext_vector_type(8)))  float    v8f;
typedef __attribute__((ext_vector_type(4)))  unsigned int u32x4;
typedef __attribute__((ext_vector_type(4)))  int i32x4;
typedef __attribute__((ext_vector_type(8)))  int i32x8;

union V16H { v16h v; v8h h[2]; };

constexpr int kB  = 16;    // batch
constexpr int kS  = 1024;  // sequence
constexpr int kH  = 1024;  // hidden (model)
constexpr int kNS = 32;    // spans
constexpr int kNH = 16;    // heads
constexpr int kNL = 3;     // labels
constexpr int kDH = 64;    // head dim
constexpr int kHL = 512;   // LSTM hidden per dir
constexpr int kG  = 2048;  // 4*kHL gate width
constexpr float kLnEps = 1e-7f;

// ------------------------------------------------------------------
// CDNA5 async global->LDS copy (ASYNCcnt path), 16B per lane.
// dsaddr = per-lane LDS byte address (low 32 bits of generic pointer).
// ------------------------------------------------------------------
__device__ __forceinline__ void async_load_b128(void* lds_ptr, const void* gptr) {
    unsigned lds_off = (unsigned)(uintptr_t)lds_ptr;
    unsigned long long ga = (unsigned long long)(uintptr_t)gptr;
    asm volatile("global_load_async_to_lds_b128 %0, %1, off"
                 :: "v"(lds_off), "v"(ga)
                 : "memory");
}

// ------------------------------------------------------------------
// CDNA5 Tensor Data Mover: 2D tile (rows x 32 f16) global -> LDS with
// hardware row padding (64B row + 16B pad == GEMM_PITCH layout).
// Wave-level op (EXEC ignored); tracked by TENSORcnt.
// D# layout per CDNA5 ISA.
// ------------------------------------------------------------------
__device__ __forceinline__ void tdm_load_tile(void* lds_ptr, const void* gptr,
                                              int rows, int Kst) {
    unsigned long long ga = (unsigned long long)(uintptr_t)gptr;
    u32x4 g0;
    g0[0] = 1u;                                     // count=1, user descriptor
    g0[1] = (unsigned)(uintptr_t)lds_ptr;           // lds_addr (bytes)
    g0[2] = (unsigned)ga;                           // global_addr[31:0]
    g0[3] = (unsigned)((ga >> 32) & 0x01ffffffull)  // global_addr[56:32]
            | (2u << 30);                           // type=2 ("image")
    i32x8 g1;
    g1[0] = (1 << 16)      // data_size = 1 (2-byte elements)
          | (1 << 20)      // pad_enable
          | (3 << 22)      // pad_interval = 3 -> every 16 DWORDs (64B row)
          | (3 << 25);     // pad_amount   = 3 -> 4 DWORDs (16B)
    g1[1] = (Kst & 0xffff) << 16;                               // tensor_dim0 lo
    g1[2] = ((Kst >> 16) & 0xffff) | ((rows & 0xffff) << 16);   // tdim0 hi | tdim1 lo
    g1[3] = (32 << 16);                                         // tdim1 hi | tile_dim0=32
    g1[4] = rows & 0xffff;                                      // tile_dim1 | tile_dim2=0
    g1[5] = Kst;                                                // tensor_dim0_stride lo
    g1[6] = 0;                                                  // stride hi | tdim1_stride lo
    g1[7] = 0;
    i32x4 gz = {0, 0, 0, 0};
#if defined(__clang_major__) && (__clang_major__ >= 23)
    i32x8 gz8 = {0, 0, 0, 0, 0, 0, 0, 0};
    __builtin_amdgcn_tensor_load_to_lds(g0, g1, gz, gz, gz8, 0);
#else
    __builtin_amdgcn_tensor_load_to_lds(g0, g1, gz, gz, 0);
#endif
}

// ------------------------------------------------------------------
// f32 -> f16 conversion (grid-stride)
// ------------------------------------------------------------------
__global__ void convert_f16_kernel(const float* __restrict__ src,
                                   _Float16* __restrict__ dst, size_t n) {
    for (size_t i = (size_t)blockIdx.x * blockDim.x + threadIdx.x; i < n;
         i += (size_t)gridDim.x * blockDim.x)
        dst[i] = (_Float16)src[i];
}

// encoder_layer [b][s][h] f32 -> x16 [s][b][h] f16 (so GEMM rows are t-major)
__global__ void pack_x_kernel(const float* __restrict__ src,
                              _Float16* __restrict__ dst) {
    const size_t n = (size_t)kB * kS * kH;
    for (size_t i = (size_t)blockIdx.x * blockDim.x + threadIdx.x; i < n;
         i += (size_t)gridDim.x * blockDim.x) {
        size_t b = i / ((size_t)kS * kH);
        size_t rem = i % ((size_t)kS * kH);
        size_t s = rem / kH, h = rem % kH;
        dst[(s * kB + b) * kH + h] = (_Float16)src[i];
    }
}

// ------------------------------------------------------------------
// WMMA GEMM:  C[M,N] = A[M,K](f16,row) * Bw[N,K](f16,row)^T + bias
// block = 256 threads = 8 waves (2x4), wave does 32x32 (2x2 WMMA tiles)
// block tile = 64(M) x 128(N); M%64==0, N%128==0, K%32==0 required.
// Double-buffered LDS: A staged via per-lane async b128 (ASYNCcnt),
// B staged via Tensor Data Mover (TENSORcnt) with HW LDS row padding.
// ------------------------------------------------------------------
#define GEMM_PITCH 40   // 32 + 8 pad (f16 elems); 40*2=80 bytes, 16B-aligned rows

__global__ __launch_bounds__(256)
void wmma_gemm_kernel(const _Float16* __restrict__ A,
                      const _Float16* __restrict__ Bw,
                      const float* __restrict__ bias,
                      float* __restrict__ Cf,      // optional f32 out
                      _Float16* __restrict__ C16,  // optional f16 out
                      int M, int N, int K) {
    __shared__ alignas(16) _Float16 At[2][64 * GEMM_PITCH];
    __shared__ alignas(16) _Float16 Bt[2][128 * GEMM_PITCH];

    const int tid   = threadIdx.x;
    const int wave  = tid >> 5;
    const int lane  = tid & 31;
    const int lrow  = lane & 15;
    const int lhalf = lane >> 4;
    const int wm = wave >> 2;   // 0..1
    const int wn = wave & 3;    // 0..3
    const int m0 = blockIdx.x * 64;
    const int n0 = blockIdx.y * 128;

    // per-thread A staging coordinates (16B granules): 64x32 tile, 1 b128/thread
    const int ar = tid >> 2, ac = (tid & 3) * 8;

    v8f acc[2][2] = {};

    auto issue_tile = [&](int buf, int k0v) {
        async_load_b128(&At[buf][ar * GEMM_PITCH + ac],
                        A + (size_t)(m0 + ar) * K + k0v + ac);
        if (wave == 0)  // one TDM descriptor moves the whole 128x32 B tile
            tdm_load_tile(&Bt[buf][0], Bw + (size_t)n0 * K + k0v, 128, K);
    };

    auto compute_tile = [&](int buf) {
        V16H af[2], bf[2];
#pragma unroll
        for (int mi = 0; mi < 2; ++mi) {
            const _Float16* p = &At[buf][(wm * 32 + mi * 16 + lrow) * GEMM_PITCH + lhalf * 8];
            af[mi].h[0] = *(const v8h*)p;
            af[mi].h[1] = *(const v8h*)(p + 16);
        }
#pragma unroll
        for (int ni = 0; ni < 2; ++ni) {
            const _Float16* p = &Bt[buf][(wn * 32 + ni * 16 + lrow) * GEMM_PITCH + lhalf * 8];
            bf[ni].h[0] = *(const v8h*)p;
            bf[ni].h[1] = *(const v8h*)(p + 16);
        }
#pragma unroll
        for (int mi = 0; mi < 2; ++mi)
#pragma unroll
            for (int ni = 0; ni < 2; ++ni)
                acc[mi][ni] = __builtin_amdgcn_wmma_f32_16x16x32_f16(
                    false, af[mi].v, false, bf[ni].v, (short)0, acc[mi][ni],
                    false, false);
    };

    issue_tile(0, 0);
    int k0 = 0;
    for (; k0 + 32 < K; k0 += 32) {          // steady state: branch-free waits
        const int buf = (k0 >> 5) & 1;
        issue_tile(buf ^ 1, k0 + 32);        // prefetch next tile
        asm volatile("s_wait_asynccnt 0x1" ::: "memory");  // this tile's A landed
        if (wave == 0) __builtin_amdgcn_s_wait_tensorcnt(1);  // this tile's B landed
        __syncthreads();                     // publish to all waves
        compute_tile(buf);
        __syncthreads();                     // reads done before buf re-issued
    }
    {                                        // peeled last tile
        const int buf = (k0 >> 5) & 1;
        asm volatile("s_wait_asynccnt 0x0" ::: "memory");
        if (wave == 0) __builtin_amdgcn_s_wait_tensorcnt(0);
        __syncthreads();
        compute_tile(buf);
    }

#pragma unroll
    for (int mi = 0; mi < 2; ++mi)
#pragma unroll
        for (int ni = 0; ni < 2; ++ni) {
            int n = n0 + wn * 32 + ni * 16 + lrow;
            float bv = bias ? bias[n] : 0.0f;
#pragma unroll
            for (int r = 0; r < 8; ++r) {
                int m = m0 + wm * 32 + mi * 16 + r + 8 * lhalf;
                float val = acc[mi][ni][r] + bv;
                if (Cf)  Cf[(size_t)m * N + n] = val;
                if (C16) C16[(size_t)m * N + n] = (_Float16)val;
            }
        }
}

// ------------------------------------------------------------------
// Persistent LSTM scan. grid = 2 blocks (dir), 1024 threads = 32 waves.
// h[16][512] f16 in LDS (WMMA A operand), c in accumulator-layout VGPRs.
// Each wave owns 16 hidden cols -> 4 gate tiles (i,f,g,o), K=512 -> 16 WMMA
// per gate tile per step.  pre = x@W_ih^T + b, precomputed f16 [t][b][2048].
// W_hh (2MB f16/dir) is L2-resident; loads feed WMMA B operands directly.
// ------------------------------------------------------------------
#define HPITCH 520   // 512 + 8 pad f16; 1040 bytes per row (16B multiple)

__global__ __launch_bounds__(1024)
void lstm_scan_kernel(const _Float16* __restrict__ pref,
                      const _Float16* __restrict__ preb,
                      const _Float16* __restrict__ whhf,
                      const _Float16* __restrict__ whhb,
                      _Float16* __restrict__ out)  // [b][s][1024]
{
    const int dir = blockIdx.x;
    const _Float16* pre = dir ? preb : pref;
    const _Float16* whh = dir ? whhb : whhf;  // [2048][512] f16
    const int outofs = dir ? kHL : 0;

    __shared__ alignas(16) _Float16 hbuf[16 * HPITCH];

    const int tid   = threadIdx.x;
    const int wave  = tid >> 5;
    const int lane  = tid & 31;
    const int lrow  = lane & 15;
    const int lhalf = lane >> 4;
    const int j0    = wave * 16;   // hidden-column base for this wave

    for (int i = tid; i < 16 * HPITCH; i += 1024) hbuf[i] = (_Float16)0;
    __syncthreads();

    v8f c = {};  // persistent cell state (16b x 16j tile, D layout)

    for (int step = 0; step < kS; ++step) {
        const int t = dir ? (kS - 1 - step) : step;
        const int tnext = dir ? (t - 1) : (t + 1);
        if (step + 1 < kS)  // prefetch next step's pre-activations into L2/L0
            __builtin_prefetch(pre + (size_t)tnext * (16 * kG) + (size_t)lane * 128, 0, 1);

        v8f acc[4] = {};
        for (int k = 0; k < kHL; k += 32) {
            V16H af;
            const _Float16* ap = &hbuf[lrow * HPITCH + k + lhalf * 8];
            af.h[0] = *(const v8h*)ap;
            af.h[1] = *(const v8h*)(ap + 16);
#pragma unroll
            for (int g = 0; g < 4; ++g) {
                V16H bf;
                const _Float16* bp =
                    whh + (size_t)(g * kHL + j0 + lrow) * kHL + k + lhalf * 8;
                bf.h[0] = *(const v8h*)bp;
                bf.h[1] = *(const v8h*)(bp + 16);
                acc[g] = __builtin_amdgcn_wmma_f32_16x16x32_f16(
                    false, af.v, false, bf.v, (short)0, acc[g], false, false);
            }
        }
        __syncthreads();  // all waves done reading hbuf

        const _Float16* pt = pre + (size_t)t * (16 * kG);
        _Float16 hnew[8];
#pragma unroll
        for (int r = 0; r < 8; ++r) {
            const int b = r + 8 * lhalf;
            const int j = j0 + lrow;
            float iv = acc[0][r] + (float)pt[b * kG + 0 * kHL + j];
            float fv = acc[1][r] + (float)pt[b * kG + 1 * kHL + j];
            float gv = acc[2][r] + (float)pt[b * kG + 2 * kHL + j];
            float ov = acc[3][r] + (float)pt[b * kG + 3 * kHL + j];
            iv = 1.0f / (1.0f + __expf(-iv));
            fv = 1.0f / (1.0f + __expf(-fv));
            ov = 1.0f / (1.0f + __expf(-ov));
            gv = tanhf(gv);
            float cv = fv * c[r] + iv * gv;
            c[r] = cv;
            float hv = ov * tanhf(cv);
            hnew[r] = (_Float16)hv;
            out[((size_t)b * kS + t) * kH + outofs + j] = (_Float16)hv;
        }
#pragma unroll
        for (int r = 0; r < 8; ++r)
            hbuf[(r + 8 * lhalf) * HPITCH + j0 + lrow] = hnew[r];
        __syncthreads();  // new h visible to everyone
    }
}

// ------------------------------------------------------------------
// Span masked-mean pooling + LayerNorm.  1 block per (b,span), 256 thr.
// ------------------------------------------------------------------
__global__ __launch_bounds__(256)
void pool_ln_kernel(const _Float16* __restrict__ lstm16,
                    const long long* __restrict__ head,
                    const long long* __restrict__ tail,
                    const float* __restrict__ gam, const float* __restrict__ bet,
                    _Float16* __restrict__ mean16, float* __restrict__ meanf) {
    const int bn  = blockIdx.x;  // b*kNS + n
    const int b   = bn / kNS;
    const int tid = threadIdx.x;
    const int c0  = tid * 4;
    const int s0  = (int)head[bn] + 1;
    const int s1  = (int)tail[bn];  // exclusive

    float v[4] = {0.f, 0.f, 0.f, 0.f};
    for (int s = s0; s < s1; ++s) {
        const _Float16* row = lstm16 + ((size_t)b * kS + s) * kH + c0;
#pragma unroll
        for (int i = 0; i < 4; ++i) v[i] += (float)row[i];
    }
    const float inv = 1.0f / (float)(s1 - s0);
#pragma unroll
    for (int i = 0; i < 4; ++i) v[i] *= inv;

    __shared__ float red[256];
    float ls = v[0] + v[1] + v[2] + v[3];
    red[tid] = ls; __syncthreads();
    for (int st = 128; st > 0; st >>= 1) {
        if (tid < st) red[tid] += red[tid + st];
        __syncthreads();
    }
    const float mu = red[0] / (float)kH;
    __syncthreads();
    float ls2 = 0.f;
#pragma unroll
    for (int i = 0; i < 4; ++i) { float d = v[i] - mu; ls2 += d * d; }
    red[tid] = ls2; __syncthreads();
    for (int st = 128; st > 0; st >>= 1) {
        if (tid < st) red[tid] += red[tid + st];
        __syncthreads();
    }
    const float rstd = rsqrtf(red[0] / (float)kH + kLnEps);
#pragma unroll
    for (int i = 0; i < 4; ++i) {
        float y = (v[i] - mu) * rstd * gam[c0 + i] + bet[c0 + i];
        meanf[(size_t)bn * kH + c0 + i]  = y;
        mean16[(size_t)bn * kH + c0 + i] = (_Float16)y;
    }
}

// ------------------------------------------------------------------
// Softmax attention, 32x32 per (b,head). 1 wave per block; thread = q row.
// ------------------------------------------------------------------
__global__ __launch_bounds__(32)
void attn_kernel(const float* __restrict__ q, const float* __restrict__ k,
                 const float* __restrict__ v, const int* __restrict__ mask,
                 _Float16* __restrict__ ctx16) {
    const int bh = blockIdx.x;           // b*kNH + h
    const int b  = bh / kNH, h = bh % kNH;
    const int r  = threadIdx.x;          // query row 0..31
    const float scale = 0.125f;          // 1/sqrt(64)

    float qr[kDH];
    const float* qp = q + ((size_t)(b * kNS) + r) * kH + h * kDH;
#pragma unroll
    for (int d = 0; d < kDH; ++d) qr[d] = qp[d];

    const bool mr = mask[b * kNS + r] != 0;
    float sc[kNS];
    float mx = -3.0e38f;
#pragma unroll
    for (int kk = 0; kk < kNS; ++kk) {
        const float* kp = k + ((size_t)(b * kNS) + kk) * kH + h * kDH;
        float s = 0.f;
#pragma unroll
        for (int d = 0; d < kDH; ++d) s += qr[d] * kp[d];
        s *= scale;
        const bool mm = mr && (mask[b * kNS + kk] != 0);
        s = mm ? s : -3.0e38f;
        sc[kk] = s;
        mx = fmaxf(mx, s);
    }
    float den = 0.f;
#pragma unroll
    for (int kk = 0; kk < kNS; ++kk) { sc[kk] = __expf(sc[kk] - mx); den += sc[kk]; }
    const float rden = 1.0f / den;
#pragma unroll
    for (int kk = 0; kk < kNS; ++kk) {
        const bool mm = mr && (mask[b * kNS + kk] != 0);
        sc[kk] = mm ? sc[kk] * rden : 0.f;
    }
    for (int d = 0; d < kDH; ++d) {
        float o = 0.f;
#pragma unroll
        for (int kk = 0; kk < kNS; ++kk)
            o += sc[kk] * v[((size_t)(b * kNS) + kk) * kH + h * kDH + d];
        ctx16[((size_t)(b * kNS) + r) * kH + h * kDH + d] = (_Float16)o;
    }
}

// ------------------------------------------------------------------
// Residual + LayerNorm + classifier.  1 block per (b,span), 256 thr.
// ------------------------------------------------------------------
__global__ __launch_bounds__(256)
void res_ln_cls_kernel(const float* __restrict__ attn,
                       const float* __restrict__ meanf,
                       const float* __restrict__ gam, const float* __restrict__ bet,
                       const float* __restrict__ clsw, const float* __restrict__ clsb,
                       float* __restrict__ out) {
    const int bn  = blockIdx.x;
    const int tid = threadIdx.x;
    const int c0  = tid * 4;

    float v[4];
#pragma unroll
    for (int i = 0; i < 4; ++i)
        v[i] = attn[(size_t)bn * kH + c0 + i] + meanf[(size_t)bn * kH + c0 + i];

    __shared__ float red[256];
    float ls = v[0] + v[1] + v[2] + v[3];
    red[tid] = ls; __syncthreads();
    for (int st = 128; st > 0; st >>= 1) {
        if (tid < st) red[tid] += red[tid + st];
        __syncthreads();
    }
    const float mu = red[0] / (float)kH;
    __syncthreads();
    float ls2 = 0.f;
#pragma unroll
    for (int i = 0; i < 4; ++i) { float d = v[i] - mu; ls2 += d * d; }
    red[tid] = ls2; __syncthreads();
    for (int st = 128; st > 0; st >>= 1) {
        if (tid < st) red[tid] += red[tid + st];
        __syncthreads();
    }
    const float rstd = rsqrtf(red[0] / (float)kH + kLnEps);
    __syncthreads();

    float p[kNL] = {0.f, 0.f, 0.f};
#pragma unroll
    for (int i = 0; i < 4; ++i) {
        float y = (v[i] - mu) * rstd * gam[c0 + i] + bet[c0 + i];
#pragma unroll
        for (int l = 0; l < kNL; ++l) p[l] += y * clsw[(size_t)l * kH + c0 + i];
    }
    for (int l = 0; l < kNL; ++l) {
        red[tid] = p[l]; __syncthreads();
        for (int st = 128; st > 0; st >>= 1) {
            if (tid < st) red[tid] += red[tid + st];
            __syncthreads();
        }
        if (tid == 0) out[(size_t)bn * kNL + l] = red[0] + clsb[l];
        __syncthreads();
    }
}

// ------------------------------------------------------------------
extern "C" void kernel_launch(void* const* d_in, const int* in_sizes, int n_in,
                              void* d_out, int out_size, void* d_ws, size_t ws_size,
                              hipStream_t stream) {
    const float* enc    = (const float*)d_in[0];
    const float* w_ih_f = (const float*)d_in[1];
    const float* w_hh_f = (const float*)d_in[2];
    const float* b_f    = (const float*)d_in[3];
    const float* w_ih_b = (const float*)d_in[4];
    const float* w_hh_b = (const float*)d_in[5];
    const float* b_b    = (const float*)d_in[6];
    const float* ln_g   = (const float*)d_in[7];
    const float* ln_b   = (const float*)d_in[8];
    const float* wq     = (const float*)d_in[9];
    const float* bq     = (const float*)d_in[10];
    const float* wk     = (const float*)d_in[11];
    const float* bk     = (const float*)d_in[12];
    const float* wv     = (const float*)d_in[13];
    const float* bv     = (const float*)d_in[14];
    const float* wo     = (const float*)d_in[15];
    const float* bo     = (const float*)d_in[16];
    const float* aln_g  = (const float*)d_in[17];
    const float* aln_b  = (const float*)d_in[18];
    const float* clsw   = (const float*)d_in[19];
    const float* clsb   = (const float*)d_in[20];
    const long long* head = (const long long*)d_in[21];
    const long long* tail = (const long long*)d_in[22];
    const int* mask       = (const int*)d_in[23];
    float* out = (float*)d_out;

    // workspace bump allocator (256B aligned)
    char* ws = (char*)d_ws;
    size_t off = 0;
    auto alloc = [&](size_t bytes) -> void* {
        void* p = ws + off;
        off += (bytes + 255) & ~(size_t)255;
        return p;
    };
    _Float16* x16    = (_Float16*)alloc((size_t)kS * kB * kH * 2);   // [s][b][h]
    _Float16* wihf16 = (_Float16*)alloc((size_t)kG * kH * 2);
    _Float16* wihb16 = (_Float16*)alloc((size_t)kG * kH * 2);
    _Float16* whhf16 = (_Float16*)alloc((size_t)kG * kHL * 2);
    _Float16* whhb16 = (_Float16*)alloc((size_t)kG * kHL * 2);
    _Float16* wq16   = (_Float16*)alloc((size_t)kH * kH * 2);
    _Float16* wk16   = (_Float16*)alloc((size_t)kH * kH * 2);
    _Float16* wv16   = (_Float16*)alloc((size_t)kH * kH * 2);
    _Float16* wo16   = (_Float16*)alloc((size_t)kH * kH * 2);
    _Float16* pref   = (_Float16*)alloc((size_t)kS * kB * kG * 2);   // [t][b][g]
    _Float16* preb   = (_Float16*)alloc((size_t)kS * kB * kG * 2);
    _Float16* lstm16 = (_Float16*)alloc((size_t)kB * kS * kH * 2);   // [b][s][h]
    _Float16* mean16 = (_Float16*)alloc((size_t)kB * kNS * kH * 2);
    float*    meanf  = (float*)alloc((size_t)kB * kNS * kH * 4);
    float*    qf     = (float*)alloc((size_t)kB * kNS * kH * 4);
    float*    kf     = (float*)alloc((size_t)kB * kNS * kH * 4);
    float*    vf     = (float*)alloc((size_t)kB * kNS * kH * 4);
    _Float16* ctx16  = (_Float16*)alloc((size_t)kB * kNS * kH * 2);
    float*    attnf  = (float*)alloc((size_t)kB * kNS * kH * 4);
    (void)ws_size; (void)in_sizes; (void)n_in; (void)out_size;

    // 1) pack / convert
    pack_x_kernel<<<4096, 256, 0, stream>>>(enc, x16);
    convert_f16_kernel<<<2048, 256, 0, stream>>>(w_ih_f, wihf16, (size_t)kG * kH);
    convert_f16_kernel<<<2048, 256, 0, stream>>>(w_ih_b, wihb16, (size_t)kG * kH);
    convert_f16_kernel<<<1024, 256, 0, stream>>>(w_hh_f, whhf16, (size_t)kG * kHL);
    convert_f16_kernel<<<1024, 256, 0, stream>>>(w_hh_b, whhb16, (size_t)kG * kHL);
    convert_f16_kernel<<<1024, 256, 0, stream>>>(wq, wq16, (size_t)kH * kH);
    convert_f16_kernel<<<1024, 256, 0, stream>>>(wk, wk16, (size_t)kH * kH);
    convert_f16_kernel<<<1024, 256, 0, stream>>>(wv, wv16, (size_t)kH * kH);
    convert_f16_kernel<<<1024, 256, 0, stream>>>(wo, wo16, (size_t)kH * kH);

    // 2) input projections: pre = x @ W_ih^T + b   (M=16384, N=2048, K=1024)
    {
        dim3 grid((kS * kB) / 64, kG / 128);
        wmma_gemm_kernel<<<grid, 256, 0, stream>>>(x16, wihf16, b_f, nullptr,
                                                   pref, kS * kB, kG, kH);
        wmma_gemm_kernel<<<grid, 256, 0, stream>>>(x16, wihb16, b_b, nullptr,
                                                   preb, kS * kB, kG, kH);
    }

    // 3) bidirectional LSTM scan (2 persistent workgroups)
    lstm_scan_kernel<<<2, 1024, 0, stream>>>(pref, preb, whhf16, whhb16, lstm16);

    // 4) span mean pooling + LayerNorm
    pool_ln_kernel<<<kB * kNS, 256, 0, stream>>>(lstm16, head, tail, ln_g, ln_b,
                                                 mean16, meanf);

    // 5) Q/K/V projections (M=512, N=1024, K=1024)
    {
        dim3 grid((kB * kNS) / 64, kH / 128);
        wmma_gemm_kernel<<<grid, 256, 0, stream>>>(mean16, wq16, bq, qf, nullptr,
                                                   kB * kNS, kH, kH);
        wmma_gemm_kernel<<<grid, 256, 0, stream>>>(mean16, wk16, bk, kf, nullptr,
                                                   kB * kNS, kH, kH);
        wmma_gemm_kernel<<<grid, 256, 0, stream>>>(mean16, wv16, bv, vf, nullptr,
                                                   kB * kNS, kH, kH);
    }

    // 6) softmax attention (32x32 per (b,head))
    attn_kernel<<<kB * kNH, 32, 0, stream>>>(qf, kf, vf, mask, ctx16);

    // 7) output projection
    {
        dim3 grid((kB * kNS) / 64, kH / 128);
        wmma_gemm_kernel<<<grid, 256, 0, stream>>>(ctx16, wo16, bo, attnf, nullptr,
                                                   kB * kNS, kH, kH);
    }

    // 8) residual + LayerNorm + classifier
    res_ln_cls_kernel<<<kB * kNS, 256, 0, stream>>>(attnf, meanf, aln_g, aln_b,
                                                    clsw, clsb, out);
}
